// AttentionNet_55456617726652
// MI455X (gfx1250) — matrix operand order repaired and checked by reading, not compiled, not run on hardware
//
#include <hip/hip_runtime.h>
#include <hip/hip_bf16.h>

// ---------------------------------------------------------------------------
// AttentionNet for MI455X (gfx1250, wave32, WMMA)
//   h = tanh(att @ Wh^T + ref @ Wv^T); scores = h @ Ws^T;
//   w = softmax(scores); out = sum_s w[s] * att[s]
// GEMM: v_wmma_f32_16x16x32_bf16 (f32 accum). A fragments staged to LDS in
// WMMA layout once per block; B pre-packed bf16 (L2-resident, 512 KB).
// ---------------------------------------------------------------------------

#define SEQ 131072
#define DIM 512

typedef __attribute__((ext_vector_type(8)))  float          v8f;
typedef __attribute__((ext_vector_type(16))) __bf16         v16bf;
typedef __attribute__((ext_vector_type(16))) unsigned short v16u;

static __device__ __forceinline__ unsigned short f2bf(float f) {
    // round-to-nearest-even f32 -> bf16
    unsigned u = __float_as_uint(f);
    unsigned r = u + 0x7FFFu + ((u >> 16) & 1u);
    return (unsigned short)(r >> 16);
}

// packed f32x2 -> bf16x2 (one v_cvt_pk_bf16_f32 when available)
static __device__ __forceinline__ unsigned pkbf(float lo, float hi) {
#if __has_builtin(__builtin_amdgcn_cvt_pk_bf16_f32)
    auto v = __builtin_amdgcn_cvt_pk_bf16_f32(lo, hi);
    return __builtin_bit_cast(unsigned, v);
#else
    return (unsigned)f2bf(lo) | ((unsigned)f2bf(hi) << 16);
#endif
}

// fast tanh: native V_TANH_F32 if exposed, else 1 - 2/(1+exp(2x))
static __device__ __forceinline__ float fast_tanh(float x) {
#if __has_builtin(__builtin_amdgcn_tanhf)
    return __builtin_amdgcn_tanhf(x);
#elif __has_builtin(__builtin_amdgcn_tanh_f32)
    return __builtin_amdgcn_tanh_f32(x);
#else
    float e = __expf(2.0f * x);
    return 1.0f - 2.0f * __builtin_amdgcn_rcpf(e + 1.0f);
#endif
}

static __device__ __forceinline__ unsigned fkey(float f) {
    unsigned u = __float_as_uint(f);
    return (u & 0x80000000u) ? ~u : (u | 0x80000000u);
}
static __device__ __forceinline__ float keyToFloat(unsigned k) {
    unsigned u = (k & 0x80000000u) ? (k ^ 0x80000000u) : ~k;
    return __uint_as_float(u);
}

// ---------------------------------------------------------------------------
__global__ void init_kernel(float* numer, float* denom, unsigned* maxkey) {
    int t = threadIdx.x;
    numer[t]       = 0.0f;
    numer[t + 256] = 0.0f;
    if (t == 0) { *denom = 0.0f; *maxkey = fkey(-INFINITY); }
}

// ---------------------------------------------------------------------------
// Pack Wh (f32 [N][K], h[s,n]=sum_k att[s,k]*Wh[n,k]) into bf16 WMMA-B tiles.
// Tile (kt,nt) = 32x16 = 256 u32 pairs. ISA B layout (32x16, 16-bit):
// lanes 0-15 hold K=0..7 & 16..23 (N=lane), lanes 16-31 hold K=8..15 & 24..31 (N=lane-16).
// u32 pair jp of lane -> K = 8*(lane>=16) + 2*jp + 8*(jp>=4) (+0,+1).
// ---------------------------------------------------------------------------
__global__ void pack_wh_bf16(const float* __restrict__ Wh,
                             unsigned* __restrict__ out) {
    int gid    = blockIdx.x * blockDim.x + threadIdx.x;   // 0..131071 pairs
    int tile   = gid >> 8;
    int within = gid & 255;
    int lane   = within >> 3;
    int jp     = within & 7;
    int kt     = tile >> 5;
    int nt     = tile & 31;
    int n      = nt * 16 + (lane & 15);
    int k      = kt * 32 + ((lane >> 4) ? 8 : 0) + 2 * jp + ((jp >= 4) ? 8 : 0);
    const float* p = Wh + (size_t)n * DIM + k;
    out[gid] = pkbf(p[0], p[1]);
}

// ---------------------------------------------------------------------------
__global__ void bias_kernel(const float* __restrict__ ref,
                            const float* __restrict__ Wv,
                            float* __restrict__ bias) {
    int d = blockIdx.x * blockDim.x + threadIdx.x;
    if (d < DIM) {
        float acc = 0.0f;
        for (int k = 0; k < DIM; ++k) acc += ref[k] * Wv[d * DIM + k];
        bias[d] = acc;
    }
}

// ---------------------------------------------------------------------------
// Main kernel. Block = 256 thr (8 wave32), 32 rows of att (2 M-tiles).
// Phase 1: cooperatively convert 32x512 f32 -> bf16 A fragments in LDS
//          (WMMA-A layout, 32 KB), once per block.
// Phase 2: wave w owns N-tiles 4w..4w+3 and both M-tiles; per K-step:
//          2 LDS A-frag loads + 4 B-frag loads, 8 WMMA (B reused x2).
//          128 v_wmma per wave total.
// Epilogue fuses tanh, *Ws, 16-lane shuffle reduce, ds_add_f32 -> scores.
// ---------------------------------------------------------------------------
__global__ __launch_bounds__(256) void attn_scores_wmma(
    const float* __restrict__ att,
    const unsigned* __restrict__ packB,   // bf16 pairs, tiled, 512 KB
    const float* __restrict__ bias,       // [512]
    const float* __restrict__ Ws,         // [512]
    float* __restrict__ scores)           // [S]
{
    __shared__ unsigned ldsA[8192];       // 32 KB: 2 fragment sets of 16 KB
    __shared__ float sarr[32];

    const int tid    = threadIdx.x;
    const int lane   = tid & 31;
    const int wave   = tid >> 5;          // 0..7
    const int hi     = lane >> 4;
    const int mcol   = lane & 15;
    const int s0     = blockIdx.x * 32;
    const int ntBase = wave * 4;

    if (tid < 32) sarr[tid] = 0.0f;

    // ---- Phase 1: stage A fragments --------------------------------------
    // task = (m 0..31, kt 0..15); thread does 2 tasks. A-frag layout (16-bit
    // 16x32): lanes<16: K 0..7 & 16..23; lanes>=16: K 8..15 & 24..31.
#pragma unroll
    for (int it = 0; it < 2; ++it) {
        int task = tid + it * 256;        // 0..511
        int m    = task & 31;
        int kt   = task >> 5;
        const float4* rp = (const float4*)(att + (size_t)(s0 + m) * DIM + kt * 32);
        float4 f0 = rp[0], f1 = rp[1], f2 = rp[2], f3 = rp[3];
        float4 f4 = rp[4], f5 = rp[5], f6 = rp[6], f7 = rp[7];
        unsigned base = (unsigned)((m >> 4) * 4096 + kt * 256);
        int mm = m & 15;
        uint4 q;
        q = make_uint4(pkbf(f0.x,f0.y), pkbf(f0.z,f0.w), pkbf(f1.x,f1.y), pkbf(f1.z,f1.w));
        *(uint4*)&ldsA[base + mm * 8] = q;                 // K 0..7
        q = make_uint4(pkbf(f2.x,f2.y), pkbf(f2.z,f2.w), pkbf(f3.x,f3.y), pkbf(f3.z,f3.w));
        *(uint4*)&ldsA[base + (mm + 16) * 8] = q;          // K 8..15
        q = make_uint4(pkbf(f4.x,f4.y), pkbf(f4.z,f4.w), pkbf(f5.x,f5.y), pkbf(f5.z,f5.w));
        *(uint4*)&ldsA[base + mm * 8 + 4] = q;             // K 16..23
        q = make_uint4(pkbf(f6.x,f6.y), pkbf(f6.z,f6.w), pkbf(f7.x,f7.y), pkbf(f7.z,f7.w));
        *(uint4*)&ldsA[base + (mm + 16) * 8 + 4] = q;      // K 24..31
    }
    __syncthreads();

    // ---- Phase 2: WMMA main loop -----------------------------------------
    v8f acc[2][4];
#pragma unroll
    for (int mt = 0; mt < 2; ++mt)
#pragma unroll
        for (int t = 0; t < 4; ++t)
#pragma unroll
            for (int r = 0; r < 8; ++r) acc[mt][t][r] = 0.0f;

#pragma unroll 2
    for (int kt = 0; kt < 16; ++kt) {
        v16u a0u = *(const v16u*)&ldsA[kt * 256 + lane * 8];
        v16u a1u = *(const v16u*)&ldsA[4096 + kt * 256 + lane * 8];
        v16bf a0 = __builtin_bit_cast(v16bf, a0u);
        v16bf a1 = __builtin_bit_cast(v16bf, a1u);
#pragma unroll
        for (int t = 0; t < 4; ++t) {
            const unsigned* bp = packB + (size_t)(kt * 32 + ntBase + t) * 256 + lane * 8;
            v16u bu = *(const v16u*)bp;
            v16bf b = __builtin_bit_cast(v16bf, bu);
            acc[0][t] = __builtin_amdgcn_wmma_f32_16x16x32_bf16(
                false, a0, false, b, (short)0, acc[0][t], false, false);
            acc[1][t] = __builtin_amdgcn_wmma_f32_16x16x32_bf16(
                false, a1, false, b, (short)0, acc[1][t], false, false);
        }
    }

    // ---- Epilogue: C layout — VGPR r, lane: row = r + 8*hi, col = lane&15.
    float rv[2][8];
#pragma unroll
    for (int mt = 0; mt < 2; ++mt)
#pragma unroll
        for (int r = 0; r < 8; ++r) rv[mt][r] = 0.0f;

#pragma unroll
    for (int t = 0; t < 4; ++t) {
        const int n    = (ntBase + t) * 16 + mcol;
        const float bv = bias[n];
        const float wv = Ws[n];
#pragma unroll
        for (int mt = 0; mt < 2; ++mt)
#pragma unroll
            for (int r = 0; r < 8; ++r)
                rv[mt][r] += fast_tanh(acc[mt][t][r] + bv) * wv;
    }
#pragma unroll
    for (int off = 8; off >= 1; off >>= 1)
#pragma unroll
        for (int mt = 0; mt < 2; ++mt)
#pragma unroll
            for (int r = 0; r < 8; ++r)
                rv[mt][r] += __shfl_xor(rv[mt][r], off, 32);
    if (mcol == 0) {
#pragma unroll
        for (int mt = 0; mt < 2; ++mt)
#pragma unroll
            for (int r = 0; r < 8; ++r)
                atomicAdd(&sarr[mt * 16 + r + 8 * hi], rv[mt][r]);  // ds_add_f32
    }
    __syncthreads();
    if (tid < 32) scores[s0 + tid] = sarr[tid];
}

// ---------------------------------------------------------------------------
__global__ __launch_bounds__(256) void max_kernel(const float* __restrict__ scores,
                                                  unsigned* __restrict__ maxkey) {
    __shared__ float wm[8];
    float mx = -INFINITY;
    for (int i = blockIdx.x * blockDim.x + threadIdx.x; i < SEQ;
         i += gridDim.x * blockDim.x)
        mx = fmaxf(mx, scores[i]);
#pragma unroll
    for (int off = 16; off >= 1; off >>= 1)
        mx = fmaxf(mx, __shfl_xor(mx, off, 32));
    if ((threadIdx.x & 31) == 0) wm[threadIdx.x >> 5] = mx;
    __syncthreads();
    if (threadIdx.x == 0) {
        float v = wm[0];
#pragma unroll
        for (int i = 1; i < 8; ++i) v = fmaxf(v, wm[i]);
        atomicMax(maxkey, fkey(v));
    }
}

// ---------------------------------------------------------------------------
__global__ __launch_bounds__(256) void wsum_kernel(
    const float* __restrict__ att,
    const float* __restrict__ scores,
    const unsigned* __restrict__ maxkey,
    float* __restrict__ numer,
    float* __restrict__ denom)
{
    __shared__ float e_lds[256];
    __shared__ float wsum[8];
    const int tid = threadIdx.x;
    const int s0  = blockIdx.x * 256;
    const float mxv = keyToFloat(*maxkey);

    float e = __expf(scores[s0 + tid] - mxv);
    e_lds[tid] = e;
    float dsum = e;
#pragma unroll
    for (int off = 16; off >= 1; off >>= 1)
        dsum += __shfl_xor(dsum, off, 32);
    if ((tid & 31) == 0) wsum[tid >> 5] = dsum;
    __syncthreads();
    if (tid == 0) {
        float t = 0.0f;
#pragma unroll
        for (int i = 0; i < 8; ++i) t += wsum[i];
        atomicAdd(denom, t);
    }

    float2 acc = make_float2(0.0f, 0.0f);
    const float2* ap = (const float2*)att + (size_t)s0 * 256 + tid;
    for (int s = 0; s < 256; ++s) {
        float2 a = ap[(size_t)s * 256];
        float ev = e_lds[s];
        acc.x += ev * a.x;
        acc.y += ev * a.y;
    }
    atomicAdd(&numer[2 * tid],     acc.x);
    atomicAdd(&numer[2 * tid + 1], acc.y);
}

__global__ void finalize_kernel(const float* __restrict__ numer,
                                const float* __restrict__ denom,
                                float* __restrict__ out) {
    int d = blockIdx.x * blockDim.x + threadIdx.x;
    if (d < DIM) out[d] = numer[d] / denom[0];
}

// ---------------------------------------------------------------------------
extern "C" void kernel_launch(void* const* d_in, const int* in_sizes, int n_in,
                              void* d_out, int out_size, void* d_ws, size_t ws_size,
                              hipStream_t stream) {
    const float* att = (const float*)d_in[0];   // [S, D]
    const float* ref = (const float*)d_in[1];   // [1, DR]
    const float* Wh  = (const float*)d_in[2];   // [D, D]
    const float* Wv  = (const float*)d_in[3];   // [D, DR]
    const float* Ws  = (const float*)d_in[4];   // [1, D]
    float* out = (float*)d_out;                 // [D]

    char* ws = (char*)d_ws;
    unsigned*  packB  = (unsigned*)ws;                       // 512 KB
    float*     bias   = (float*)(ws + 524288);               // 2 KB
    float*     scores = (float*)(ws + 526336);               // 512 KB
    float*     numer  = (float*)(ws + 1050624);              // 2 KB
    float*     denom  = (float*)(ws + 1052672);              // 4 B
    unsigned*  maxkey = (unsigned*)(ws + 1052676);           // 4 B

    init_kernel<<<1, 256, 0, stream>>>(numer, denom, maxkey);
    pack_wh_bf16<<<512, 256, 0, stream>>>(Wh, packB);
    bias_kernel<<<2, 256, 0, stream>>>(ref, Wv, bias);
    attn_scores_wmma<<<SEQ / 32, 256, 0, stream>>>(att, packB, bias, Ws, scores);
    max_kernel<<<256, 256, 0, stream>>>(scores, maxkey);
    wsum_kernel<<<SEQ / 256, 256, 0, stream>>>(att, scores, maxkey, numer, denom);
    finalize_kernel<<<2, 256, 0, stream>>>(numer, denom, out);
}
